// HybridSurvivalQ_v3_9509057593684
// MI455X (gfx1250) — matrix operand order repaired and checked by reading, not compiled, not run on hardware
//
#include <hip/hip_runtime.h>
#include <hip/hip_bf16.h>

typedef __attribute__((ext_vector_type(16))) _Float16 v16h;
typedef __attribute__((ext_vector_type(8)))  float    v8f;

#define NQ 7
#define QDIM 128

// ---------------------------------------------------------------------------
// Workspace layout (in _Float16 units). Weights are stored pre-swizzled into
// WMMA B-fragment order: index = ((nt*(K/32)+kb)*32 + lane)*16 + e  with
//   n = nt*16 + lane%16 ,  k = kb*32 + 16*(e>=8) + 8*(lane>=16) + e%8
// so each lane's 16-half fragment is one contiguous, 32B-aligned vector.
//   [0)            q_out          32768*14
//   [458752)       enc_w1  swz    4096   (K=64 , N=64)
//   [462848)       enc_w2  swz    2048   (K=64 , N=32)
//   [464896)       fus_w1  swz    16384  (K=128 padded from 110, N=128)
//   [481280)       fus_w2  swz    8192   (K=128, N=64)
//   [489472)       fus_w3  swz    2048   (K=64 , N=32)
// total 491520 halves = 983040 bytes
// ---------------------------------------------------------------------------
#define WS_QOUT 0
#define WS_W1   458752
#define WS_W2   462848
#define WS_F1   464896
#define WS_F2   481280
#define WS_F3   489472

// ===========================================================================
// Weight f32 -> f16 swizzled staging
// ===========================================================================
__device__ __forceinline__ void swizzle_weight(
    const float* __restrict__ W, int K, int Ksrc, int N,
    _Float16* __restrict__ dst, int t, int stride)
{
    int nkb   = K >> 5;
    int total = (N >> 4) * nkb * 32 * 16;
    for (int i = t; i < total; i += stride) {
        int e    = i & 15;
        int L    = (i >> 4) & 31;
        int rest = i >> 9;
        int kb   = rest % nkb;
        int nt   = rest / nkb;
        int k    = (kb << 5) + ((e >> 3) << 4) + ((L >> 4) << 3) + (e & 7);
        int n    = (nt << 4) + (L & 15);
        float v  = (k < Ksrc) ? W[(size_t)k * N + n] : 0.0f;
        dst[i]   = (_Float16)v;
    }
}

__global__ __launch_bounds__(256) void prep_kernel(
    const float* __restrict__ enc_w1, const float* __restrict__ enc_w2,
    const float* __restrict__ fus_w1, const float* __restrict__ fus_w2,
    const float* __restrict__ fus_w3, _Float16* __restrict__ ws)
{
    int t = threadIdx.x + blockIdx.x * blockDim.x;
    int stride = blockDim.x * gridDim.x;
    swizzle_weight(enc_w1,  64,  64,  64, ws + WS_W1, t, stride);
    swizzle_weight(enc_w2,  64,  64,  32, ws + WS_W2, t, stride);
    swizzle_weight(fus_w1, 128, 110, 128, ws + WS_F1, t, stride);
    swizzle_weight(fus_w2, 128, 128,  64, ws + WS_F2, t, stride);
    swizzle_weight(fus_w3,  64,  64,  32, ws + WS_F3, t, stride);
}

// ===========================================================================
// Quantum circuit: one wave32 per batch row, 4 complex amplitudes per lane.
// Amplitude index k = lane*4 + j. Qubit q acts on bit (6-q) of k:
//   bits 6..2 -> cross-lane (shfl_xor), bits 1..0 -> in-register.
// ===========================================================================
__device__ __forceinline__ void apply_ry(float re[4], float im[4],
                                         float th, int q, int lane)
{
    float s, c;
    __sincosf(0.5f * th, &s, &c);
    int bp = 6 - q;
    if (bp >= 2) {
        int mask = 1 << (bp - 2);
        int bit  = (lane >> (bp - 2)) & 1;
        float sg = bit ? s : -s;     // a0' = c*a0 - s*a1 ; a1' = s*a0 + c*a1
#pragma unroll
        for (int j = 0; j < 4; ++j) {
            float pr = __shfl_xor(re[j], mask);
            float pi = __shfl_xor(im[j], mask);
            re[j] = c * re[j] + sg * pr;
            im[j] = c * im[j] + sg * pi;
        }
    } else {
        int m = 1 << bp;             // 1 or 2
#pragma unroll
        for (int j0 = 0; j0 < 4; ++j0) {
            if (j0 & m) continue;
            int j1 = j0 | m;
            float a0r = re[j0], a0i = im[j0], a1r = re[j1], a1i = im[j1];
            re[j0] = c * a0r - s * a1r;  im[j0] = c * a0i - s * a1i;
            re[j1] = s * a0r + c * a1r;  im[j1] = s * a0i + c * a1i;
        }
    }
}

__device__ __forceinline__ void apply_rz(float re[4], float im[4],
                                         float th, int q, int lane)
{
    float s, c;
    __sincosf(0.5f * th, &s, &c);
    int bp = 6 - q;
#pragma unroll
    for (int j = 0; j < 4; ++j) {
        int k   = (lane << 2) | j;
        int bit = (k >> bp) & 1;
        float ss = bit ? -s : s;     // exp(-i th/2) on bit0, conj on bit1
        float r = re[j], i2 = im[j];
        re[j] = r * c + i2 * ss;
        im[j] = i2 * c - r * ss;
    }
}

__global__ __launch_bounds__(256) void qsim_kernel(
    const float* __restrict__ x_q, const float* __restrict__ scales,
    const float* __restrict__ biases, const float* __restrict__ wry,
    const float* __restrict__ wrz, _Float16* __restrict__ qout)
{
    __shared__ float lds[8][2 * QDIM];   // per-wave scratch for CNOT gathers
    int lane = threadIdx.x & 31;
    // SGPR-uniform wave id: keeps LDS addressing scalar.
    int wave = __builtin_amdgcn_readfirstlane(threadIdx.x) >> 5;
    int row  = blockIdx.x * 8 + wave;

    float re[4], im[4];
#pragma unroll
    for (int j = 0; j < 4; ++j) { re[j] = 0.0f; im[j] = 0.0f; }
    if (lane == 0) re[0] = 1.0f;         // |0...0>

    float ang[NQ];
#pragma unroll
    for (int i = 0; i < NQ; ++i)
        ang[i] = scales[i] * x_q[(size_t)row * NQ + i] + biases[i];

    for (int layer = 0; layer < 3; ++layer) {
#pragma unroll
        for (int i = 0; i < NQ; ++i) apply_ry(re, im, ang[i], i, lane);
#pragma unroll
        for (int i = 0; i < NQ; ++i) {
            apply_ry(re, im, wry[layer * NQ + i], i, lane);
            apply_rz(re, im, wrz[layer * NQ + i], i, lane);
        }
        int shift = layer + 1;
#pragma unroll
        for (int i = 0; i < NQ; ++i) {
            int c = i, t = (i + shift) % NQ;
            // state'[k] = state[ k ^ (bit_c(k) << (6-t)) ] — gather via LDS.
            // Same-wave LDS ops are processed in order (ISA §7.3): no barrier.
#pragma unroll
            for (int j = 0; j < 4; ++j) {
                int k = (lane << 2) | j;
                lds[wave][k]        = re[j];
                lds[wave][QDIM + k] = im[j];
            }
#pragma unroll
            for (int j = 0; j < 4; ++j) {
                int k  = (lane << 2) | j;
                int kp = k ^ ((((k >> (6 - c)) & 1)) << (6 - t));
                re[j] = lds[wave][kp];
                im[j] = lds[wave][QDIM + kp];
            }
        }
    }

    // 14 features: probs @ SZ^T (7) and probs @ SZZ^T (7)
    float f[14];
#pragma unroll
    for (int i = 0; i < 14; ++i) f[i] = 0.0f;
#pragma unroll
    for (int j = 0; j < 4; ++j) {
        int k = (lane << 2) | j;
        float p = re[j] * re[j] + im[j] * im[j];
#pragma unroll
        for (int i = 0; i < 7; ++i) {
            float sz  = ((k >> (6 - i)) & 1) ? -1.0f : 1.0f;
            float szn = ((k >> (6 - ((i + 1) % 7))) & 1) ? -1.0f : 1.0f;
            f[i]     += p * sz;
            f[7 + i] += p * sz * szn;
        }
    }
#pragma unroll
    for (int i = 0; i < 14; ++i) {
#pragma unroll
        for (int off = 16; off >= 1; off >>= 1)
            f[i] += __shfl_xor(f[i], off);
    }
    if (lane == 0) {
#pragma unroll
        for (int i = 0; i < 14; ++i)
            qout[(size_t)row * 14 + i] = (_Float16)f[i];
    }
}

// ===========================================================================
// WMMA fragment helpers (CDNA5 16x16x32 f16, wave32 layouts per ISA §7.12.2)
// ===========================================================================
__device__ __forceinline__ v16h load_A(const _Float16* A, int lda,
                                       int row0, int k0, int lane)
{
    // A 16x32: lane L holds row M=L%16; halves e<8 -> K = e + 8*(L>=16),
    // halves e>=8 -> K = 16 + (e-8) + 8*(L>=16).
    int m  = lane & 15;
    int kb = (lane >> 4) << 3;
    const _Float16* p = A + (size_t)(row0 + m) * lda + k0;
    v16h a;
#pragma unroll
    for (int e = 0; e < 8; ++e) a[e] = p[kb + e];
#pragma unroll
    for (int e = 0; e < 8; ++e) a[e + 8] = p[16 + kb + e];
    return a;
}

// One logical GEMM over a 64-row LDS tile: 8 waves split the 16x16 output
// tiles; K in steps of 32 per WMMA; bias + ReLU fused into the C store.
// B comes pre-swizzled: one contiguous 32B vector per lane per fragment.
__device__ __forceinline__ void gemm_tiles(const _Float16* A, int lda,
                                           const _Float16* __restrict__ Bsw,
                                           int K, int N,
                                           const float* __restrict__ bias,
                                           _Float16* C, int ldc,
                                           int lane, int wave)
{
    int nkb    = K >> 5;
    int ntn    = N >> 4;
    int ntiles = 4 * ntn;                 // 4 row strips x (N/16) col strips
    for (int t = wave; t < ntiles; t += 8) {   // wave-uniform scalar loop
        int rs = t / ntn, nt = t % ntn;
        int r0 = rs << 4, n0 = nt << 4;
        v8f acc = {};
        for (int kb = 0; kb < nkb; ++kb) {
            v16h a = load_A(A, lda, r0, kb << 5, lane);
            v16h b = *(const v16h*)(Bsw + ((size_t)(nt * nkb + kb) * 32 + lane) * 16);
            acc = __builtin_amdgcn_wmma_f32_16x16x32_f16(
                false, a, false, b, (short)0, acc, false, false);
        }
        // C/D: reg r -> M = r + 8*(lane>=16), N = lane%16
        int n  = lane & 15;
        int mb = (lane >> 4) << 3;
        float bb = bias[n0 + n];
#pragma unroll
        for (int r = 0; r < 8; ++r) {
            float v = acc[r] + bb;
            v = v > 0.0f ? v : 0.0f;
            C[(size_t)(r0 + mb + r) * ldc + n0 + n] = (_Float16)v;
        }
    }
}

// ===========================================================================
// Fused MLP: 64 rows per block, all six layers LDS-resident.
// comb layout: [0,14) q_out | [14,46) c_enc | [46,110) x_c | [110,128) zeros
// ===========================================================================
__global__ __launch_bounds__(256) void mlp_kernel(
    const float* __restrict__ x_c, const _Float16* __restrict__ ws,
    const float* __restrict__ enc_b1, const float* __restrict__ enc_b2,
    const float* __restrict__ fus_b1, const float* __restrict__ fus_b2,
    const float* __restrict__ fus_b3, const float* __restrict__ fus_w4,
    const float* __restrict__ fus_b4, float* __restrict__ out)
{
    __shared__ _Float16 comb[64 * 128];   // 16 KB
    __shared__ _Float16 h1  [64 * 64];    //  8 KB
    __shared__ _Float16 hbig[64 * 128];   // 16 KB
    __shared__ _Float16 h3  [64 * 32];    //  4 KB

    const _Float16* qout = ws + WS_QOUT;

    int tid  = threadIdx.x;
    int lane = tid & 31;
    int wave = __builtin_amdgcn_readfirstlane(tid) >> 5;  // SGPR-uniform
    int row0 = blockIdx.x * 64;

    // Stage x_c (vectorized float4), q_out, zeros into comb.
    for (int idx = tid; idx < 64 * 16; idx += 256) {       // x_c: 64 rows x 16 float4
        int r = idx >> 4, c4 = idx & 15;
        float4 v = ((const float4*)(x_c + (size_t)(row0 + r) * 64))[c4];
        _Float16* d = comb + r * 128 + 46 + c4 * 4;
        d[0] = (_Float16)v.x; d[1] = (_Float16)v.y;
        d[2] = (_Float16)v.z; d[3] = (_Float16)v.w;
    }
    for (int idx = tid; idx < 64 * 14; idx += 256) {       // q_out (already f16)
        int r = idx / 14, c = idx % 14;
        comb[r * 128 + c] = qout[(size_t)(row0 + r) * 14 + c];
    }
    for (int idx = tid; idx < 64 * 50; idx += 256) {       // zero c_enc slot + K pad
        int r = idx / 50, c = idx % 50;
        comb[r * 128 + ((c < 32) ? (14 + c) : (110 + c - 32))] = (_Float16)0.0f;
    }
    __syncthreads();

    // enc layer 1: h1 = relu(x_c @ enc_w1 + b1)                [64x64, K=64]
    gemm_tiles(comb + 46, 128, ws + WS_W1, 64, 64, enc_b1, h1, 64, lane, wave);
    __syncthreads();
    // enc layer 2: c_enc = relu(h1 @ enc_w2 + b2) -> comb[:,14:46)  [64x32, K=64]
    gemm_tiles(h1, 64, ws + WS_W2, 64, 32, enc_b2, comb + 14, 128, lane, wave);
    __syncthreads();
    // fusion 1: hbig = relu(comb @ fus_w1 + b1)                [64x128, K=128]
    gemm_tiles(comb, 128, ws + WS_F1, 128, 128, fus_b1, hbig, 128, lane, wave);
    __syncthreads();
    // fusion 2: h1 = relu(hbig @ fus_w2 + b2)                  [64x64, K=128]
    gemm_tiles(hbig, 128, ws + WS_F2, 128, 64, fus_b2, h1, 64, lane, wave);
    __syncthreads();
    // fusion 3: h3 = relu(h1 @ fus_w3 + b3)                    [64x32, K=64]
    gemm_tiles(h1, 64, ws + WS_F3, 64, 32, fus_b3, h3, 32, lane, wave);
    __syncthreads();

    // fusion 4: out = h3 @ fus_w4 + b4   (N=1 -> plain VALU dot per row)
    if (tid < 64) {
        float acc = fus_b4[0];
#pragma unroll
        for (int k = 0; k < 32; ++k)
            acc += (float)h3[tid * 32 + k] * fus_w4[k];
        out[row0 + tid] = acc;
    }
}

// ===========================================================================
extern "C" void kernel_launch(void* const* d_in, const int* in_sizes, int n_in,
                              void* d_out, int out_size, void* d_ws, size_t ws_size,
                              hipStream_t stream)
{
    const float* x_q    = (const float*)d_in[0];
    const float* x_c    = (const float*)d_in[1];
    const float* scales = (const float*)d_in[2];
    const float* biases = (const float*)d_in[3];
    const float* wry    = (const float*)d_in[4];
    const float* wrz    = (const float*)d_in[5];
    const float* enc_w1 = (const float*)d_in[6];
    const float* enc_b1 = (const float*)d_in[7];
    const float* enc_w2 = (const float*)d_in[8];
    const float* enc_b2 = (const float*)d_in[9];
    const float* fus_w1 = (const float*)d_in[10];
    const float* fus_b1 = (const float*)d_in[11];
    const float* fus_w2 = (const float*)d_in[12];
    const float* fus_b2 = (const float*)d_in[13];
    const float* fus_w3 = (const float*)d_in[14];
    const float* fus_b3 = (const float*)d_in[15];
    const float* fus_w4 = (const float*)d_in[16];
    const float* fus_b4 = (const float*)d_in[17];

    _Float16* ws = (_Float16*)d_ws;
    float* out   = (float*)d_out;

    prep_kernel<<<64, 256, 0, stream>>>(enc_w1, enc_w2, fus_w1, fus_w2, fus_w3, ws);
    qsim_kernel<<<4096, 256, 0, stream>>>(x_q, scales, biases, wry, wrz, ws + WS_QOUT);
    mlp_kernel<<<512, 256, 0, stream>>>(x_c, ws, enc_b1, enc_b2,
                                        fus_b1, fus_b2, fus_b3,
                                        fus_w4, fus_b4, out);
}
